// MultiHeadDotProductAttention_42846593744913
// MI455X (gfx1250) — matrix operand order repaired and checked by reading, not compiled, not run on hardware
//
#include <hip/hip_runtime.h>
#include <hip/hip_bf16.h>

// ---------------------------------------------------------------------------
// MultiHeadDotProductAttention with QK-LayerNorm for MI455X (gfx1250, wave32)
// bf16 WMMA (v_wmma_f32_16x16x32_bf16) for all matmuls, flash-attention
// streaming softmax (B*H*T*T scores never materialized), and Tensor Data
// Mover (tensor_load_to_lds + s_wait_tensorcnt) double-buffering the K tiles.
// ---------------------------------------------------------------------------

typedef __bf16 bf16_t;
typedef __bf16 v16bf __attribute__((ext_vector_type(16)));
typedef __bf16 v8bf  __attribute__((ext_vector_type(8)));
typedef float  v8f   __attribute__((ext_vector_type(8)));

#define BB   2
#define TT   2048
#define EE   1024
#define HH   16
#define DD   64
#define NHD  1024
#define LN_EPS 1e-5f

#if defined(__gfx1250__) && __has_builtin(__builtin_amdgcn_tensor_load_to_lds) && \
    __has_builtin(__builtin_amdgcn_s_wait_tensorcnt)
#define HAVE_TDM 1
#else
#define HAVE_TDM 0
#endif

__device__ __forceinline__ v16bf bf_combine(v8bf lo, v8bf hi) {
  v16bf r;
#pragma unroll
  for (int i = 0; i < 8; ++i) { r[i] = lo[i]; r[i + 8] = hi[i]; }
  return r;
}

#if HAVE_TDM
typedef unsigned int u32x4 __attribute__((ext_vector_type(4)));
typedef int          i32x4 __attribute__((ext_vector_type(4)));
typedef int          i32x8 __attribute__((ext_vector_type(8)));

// TDM 2D tile load: 64 rows x 64 bf16, row stride NHD elements, into LDS.
// D# bit layout per CDNA5 ISA (groups 0/1; 2D -> remaining
// groups zero). lds byte address = low 32 bits of the generic shared pointer
// This toolchain (clang-23 / therock) declares the 6-arg builtin form:
//   (u32x4 g0, i32x8 g1, i32x4, i32x4, i32x8, i32 cpol)
__device__ __forceinline__ void tdm_load_ktile(const bf16_t* gsrc, bf16_t* ldst) {
  const unsigned long long ga = (unsigned long long)(uintptr_t)gsrc;
  const unsigned lds_off = (unsigned)(uintptr_t)ldst;
  u32x4 g0;
  g0[0] = 1u;                                            // count=1 (valid, user mode)
  g0[1] = lds_off;                                       // lds_addr (bytes)
  g0[2] = (unsigned)ga;                                  // global_addr[31:0]
  g0[3] = ((unsigned)(ga >> 32) & 0x01ffffffu)           // global_addr[56:32]
          | (2u << 30);                                  // type=2 ("image")
  i32x8 g1;
  g1[0] = 1 << 16;                                       // data_size=1 -> 2 bytes
  g1[1] = (int)(((unsigned)NHD & 0xffffu) << 16);        // tensor_dim0 = NHD
  g1[2] = (int)(((unsigned)TT & 0xffffu) << 16);         // tensor_dim1 = TT
  g1[3] = (int)(64u << 16);                              // tile_dim0 = 64
  g1[4] = 64;                                            // tile_dim1 = 64, tile_dim2 = 0
  g1[5] = NHD;                                           // tensor_dim0_stride lo32
  g1[6] = 0;                                             // stride0 hi / stride1 lo
  g1[7] = 0;
  const i32x4 gz4 = {0, 0, 0, 0};
  const i32x8 gz8 = {0, 0, 0, 0, 0, 0, 0, 0};
  __builtin_amdgcn_tensor_load_to_lds(g0, g1, gz4, gz4, gz8, 0);
}
#endif

// ---------------------------------------------------------------------------
// f32 -> bf16 bulk convert (4 elements / thread)
// ---------------------------------------------------------------------------
__global__ __launch_bounds__(256)
void cvt_f32_to_bf16(const float* __restrict__ x, bf16_t* __restrict__ y, int n4) {
  int i = blockIdx.x * 256 + threadIdx.x;
  if (i >= n4) return;
  float4 v = ((const float4*)x)[i];
  union { bf16_t b[4]; unsigned long long u; } o;
  o.b[0] = (bf16_t)v.x; o.b[1] = (bf16_t)v.y;
  o.b[2] = (bf16_t)v.z; o.b[3] = (bf16_t)v.w;
  ((unsigned long long*)y)[i] = o.u;
}

// ---------------------------------------------------------------------------
// C[M,N] = A[M,K] * W[N,K]^T   (bf16 inputs, f32 accumulate)
// Each wave computes a 16x64 strip of C: 4x v_wmma_f32_16x16x32_bf16 per
// 32-deep K step, A fragment reused across the 4 N tiles.
// Block = 128 threads (4 waves stacked in M) -> 64x64 block tile.
// ---------------------------------------------------------------------------
template <bool OUT_BF16>
__global__ __launch_bounds__(128)
void gemm_bf16_wmma(const bf16_t* __restrict__ A,  // [M,K] row-major
                    const bf16_t* __restrict__ W,  // [N,K] row-major (torch Linear)
                    void* __restrict__ Cout,       // [M,N]
                    int M, int N, int K) {
  const int lane   = threadIdx.x & 31;
  const int wave   = threadIdx.x >> 5;
  const int lanelo = lane & 15;
  const int hi     = lane >> 4;

  const int m0 = blockIdx.x * 64 + wave * 16;
  const int n0 = blockIdx.y * 64;

  // A fragment: lane half selects K sub-range {0..7,16..23} / {8..15,24..31}
  const bf16_t* arow = A + (size_t)(m0 + lanelo) * K + hi * 8;
  // B fragment: lane = N column, lane half selects K sub-range {0..7,16..23} / {8..15,24..31}
  const bf16_t* wrow = W + (size_t)(n0 + lanelo) * K + hi * 8;

  v8f acc[4] = {};
  for (int k = 0; k < K; k += 32) {
    v8bf alo = *(const v8bf*)(arow + k);
    v8bf ahi = *(const v8bf*)(arow + k + 16);
    v16bf a = bf_combine(alo, ahi);
#pragma unroll
    for (int t = 0; t < 4; ++t) {
      const bf16_t* wp = wrow + (size_t)t * 16 * K + k;
      v16bf b = bf_combine(*(const v8bf*)(wp), *(const v8bf*)(wp + 16));
      acc[t] = __builtin_amdgcn_wmma_f32_16x16x32_bf16(
          false, a, false, b, (short)0, acc[t], false, false);
    }
    if (k + 32 < K) {  // global_prefetch_b8 for the next K chunk
      __builtin_prefetch(arow + k + 32, 0, 3);
      __builtin_prefetch(wrow + k + 32, 0, 3);
    }
  }

  // C layout: lanes 0-15 hold rows m0+0..7, lanes 16-31 rows m0+8..15
  const int row_base = m0 + hi * 8;
#pragma unroll
  for (int t = 0; t < 4; ++t) {
    const int col = n0 + t * 16 + lanelo;
#pragma unroll
    for (int r = 0; r < 8; ++r) {
      const size_t idx = (size_t)(row_base + r) * N + col;
      if (OUT_BF16) ((bf16_t*)Cout)[idx] = (bf16_t)acc[t][r];
      else          ((float*)Cout)[idx]  = acc[t][r];
    }
  }
}

// ---------------------------------------------------------------------------
// Row LayerNorm over NH=1024, f32 in -> bf16 out. One 256-thread block / row.
// ---------------------------------------------------------------------------
__global__ __launch_bounds__(256)
void layernorm_to_bf16(const float* __restrict__ x,      // [R, NHD]
                       const float* __restrict__ scale,  // [NHD]
                       const float* __restrict__ bias,   // [NHD]
                       bf16_t* __restrict__ y) {         // [R, NHD]
  const int row = blockIdx.x;
  const float* xr = x + (size_t)row * NHD;

  float v[4], lsum = 0.f, lsq = 0.f;
#pragma unroll
  for (int i = 0; i < 4; ++i) {
    v[i] = xr[threadIdx.x + i * 256];
    lsum += v[i];
    lsq  += v[i] * v[i];
  }
#pragma unroll
  for (int sh = 16; sh >= 1; sh >>= 1) {
    lsum += __shfl_xor(lsum, sh, 32);
    lsq  += __shfl_xor(lsq,  sh, 32);
  }
  __shared__ float s_sum[8], s_sq[8];
  if ((threadIdx.x & 31) == 0) {
    s_sum[threadIdx.x >> 5] = lsum;
    s_sq [threadIdx.x >> 5] = lsq;
  }
  __syncthreads();
  float tsum = 0.f, tsq = 0.f;
#pragma unroll
  for (int w = 0; w < 8; ++w) { tsum += s_sum[w]; tsq += s_sq[w]; }

  const float mu   = tsum * (1.0f / NHD);
  const float var  = tsq * (1.0f / NHD) - mu * mu;
  const float rstd = rsqrtf(var + LN_EPS);

  bf16_t* yr = y + (size_t)row * NHD;
#pragma unroll
  for (int i = 0; i < 4; ++i) {
    const int c = threadIdx.x + i * 256;
    yr[c] = (bf16_t)((v[i] - mu) * rstd * scale[c] + bias[c]);
  }
}

// ---------------------------------------------------------------------------
// Causal flash attention per (b,h). Grid = (B*H, T/64); block = 128 (4 waves).
// Each wave owns 16 query rows; KV streamed in 64-row tiles.
// K tiles: TDM (tensor_load_to_lds) double-buffered, issued one tile ahead so
// the DMA overlaps the V transpose + WMMA compute; wave 0 drains TENSORcnt to
// <=1 before the workgroup barrier.
// S = Q K^T via 8 WMMAs / tile; P converted C-layout -> A-fragment through
// per-wave LDS; O += P V via 8 WMMAs / tile.
// ---------------------------------------------------------------------------
__global__ __launch_bounds__(128)
void flash_attn_wmma(const bf16_t* __restrict__ Qn,  // [B,T,NHD] (LN'd)
                     const bf16_t* __restrict__ Kn,  // [B,T,NHD] (LN'd)
                     const bf16_t* __restrict__ Vb,  // [B,T,NHD]
                     bf16_t* __restrict__ Ao) {      // [B,T,NHD]
  const int lane   = threadIdx.x & 31;
  const int wave   = threadIdx.x >> 5;
  const int lanelo = lane & 15;
  const int hi     = lane >> 4;

  const int bh = blockIdx.x;
  const int b  = bh / HH;
  const int h  = bh - b * HH;
  const int qt = blockIdx.y;            // 64-row query tile
  const int m0 = qt * 64 + wave * 16;   // this wave's query rows

  const size_t base = (size_t)b * TT * NHD + (size_t)h * DD;

  __shared__ bf16_t ldsK[2][64 * 64];    // K tile double buffer, row-major [j][d]
  __shared__ bf16_t ldsVt[64 * 80];      // V tile transposed [d][j], padded
  __shared__ bf16_t ldsP[4][16 * 64];    // per-wave P staging, row-major [m][j]

  // Preload Q fragments for both 32-deep D chunks (D=64)
  v16bf aq[2];
  {
    const bf16_t* qrow = Qn + base + (size_t)(m0 + lanelo) * NHD + hi * 8;
#pragma unroll
    for (int kc = 0; kc < 2; ++kc)
      aq[kc] = bf_combine(*(const v8bf*)(qrow + kc * 32),
                          *(const v8bf*)(qrow + kc * 32 + 16));
  }

  v8f o[4] = {};                     // O tile: 16 rows x 64 d
  float mrow[8], lrow[8];
#pragma unroll
  for (int r = 0; r < 8; ++r) { mrow[r] = -1e30f; lrow[r] = 0.f; }

  const float sm_scale = 0.125f;     // 1/sqrt(D), D=64

#if HAVE_TDM
  if (wave == 0) tdm_load_ktile(Kn + base, &ldsK[0][0]);   // prefetch tile 0
#endif

  for (int jt = 0; jt <= qt; ++jt) {
    const int cur = jt & 1;
    const int j0  = jt * 64;

#if !HAVE_TDM
    // Fallback cooperative K stage (row-major copy)
#pragma unroll
    for (int c = 0; c < 4; ++c) {
      const int chunk = threadIdx.x + c * 128;   // 512 chunks of 8 bf16
      const int rr = chunk >> 3;
      const int cc = (chunk & 7) * 8;
      *(v8bf*)(&ldsK[cur][rr * 64 + cc]) =
          *(const v8bf*)(Kn + base + (size_t)(j0 + rr) * NHD + cc);
    }
#endif
    // V tile transpose stage (TDM cannot transpose): [j][d] -> [d][j]
#pragma unroll
    for (int c = 0; c < 32; ++c) {
      const int idx = threadIdx.x + c * 128;     // 4096 scalar elements
      const int rr = idx >> 6;                   // j within tile
      const int dd = idx & 63;                   // d
      ldsVt[dd * 80 + rr] = Vb[base + (size_t)(j0 + rr) * NHD + dd];
    }

#if HAVE_TDM
    if (wave == 0) {
      if (jt < qt) {
        // issue next K tile into the other buffer, keep current in flight <=1
        tdm_load_ktile(Kn + base + (size_t)(j0 + 64) * NHD, &ldsK[cur ^ 1][0]);
        __builtin_amdgcn_s_wait_tensorcnt(1);
      } else {
        __builtin_amdgcn_s_wait_tensorcnt(0);
      }
    }
#endif
    __syncthreads();

    // ---- S = Q K^T (4 N-tiles x 2 K-steps) + causal mask + scale ----
    v8f s[4];
    float mnew[8];
#pragma unroll
    for (int r = 0; r < 8; ++r) mnew[r] = mrow[r];
#pragma unroll
    for (int nt = 0; nt < 4; ++nt) {
      v8f sa = {};
#pragma unroll
      for (int kc = 0; kc < 2; ++kc) {
        const bf16_t* kp = &ldsK[cur][(nt * 16 + lanelo) * 64 + kc * 32 + hi * 8];
        v16bf bk = bf_combine(*(const v8bf*)(kp), *(const v8bf*)(kp + 16));
        sa = __builtin_amdgcn_wmma_f32_16x16x32_bf16(
            false, aq[kc], false, bk, (short)0, sa, false, false);
      }
      const int j = j0 + nt * 16 + lanelo;
#pragma unroll
      for (int r = 0; r < 8; ++r) {
        const int m = m0 + hi * 8 + r;
        float sv = sa[r] * sm_scale;
        sv = (j <= m) ? sv : -1e30f;            // causal mask
        sa[r] = sv;
        mnew[r] = fmaxf(mnew[r], sv);
      }
      s[nt] = sa;
    }
    // row-max across the 16 lanes holding this row (stays in half-wave)
#pragma unroll
    for (int sh = 8; sh >= 1; sh >>= 1)
#pragma unroll
      for (int r = 0; r < 8; ++r)
        mnew[r] = fmaxf(mnew[r], __shfl_xor(mnew[r], sh, 32));

    // ---- online-softmax rescale ----
    float alpha[8];
#pragma unroll
    for (int r = 0; r < 8; ++r) {
      alpha[r] = __expf(mrow[r] - mnew[r]);
      mrow[r] = mnew[r];
    }
#pragma unroll
    for (int nt = 0; nt < 4; ++nt)
#pragma unroll
      for (int r = 0; r < 8; ++r) o[nt][r] *= alpha[r];

    float lacc[8];
#pragma unroll
    for (int r = 0; r < 8; ++r) lacc[r] = 0.f;
#pragma unroll
    for (int nt = 0; nt < 4; ++nt)
#pragma unroll
      for (int r = 0; r < 8; ++r) {
        const float pv = __expf(s[nt][r] - mrow[r]);
        s[nt][r] = pv;
        lacc[r] += pv;
      }
#pragma unroll
    for (int sh = 8; sh >= 1; sh >>= 1)
#pragma unroll
      for (int r = 0; r < 8; ++r)
        lacc[r] += __shfl_xor(lacc[r], sh, 32);
#pragma unroll
    for (int r = 0; r < 8; ++r) lrow[r] = lrow[r] * alpha[r] + lacc[r];

    // ---- P: C-layout -> bf16 A-fragment via per-wave LDS ----
#pragma unroll
    for (int nt = 0; nt < 4; ++nt)
#pragma unroll
      for (int r = 0; r < 8; ++r)
        ldsP[wave][(hi * 8 + r) * 64 + nt * 16 + lanelo] = (bf16_t)s[nt][r];

    // ---- O += P V (2 K-steps over j x 4 d-tiles) ----
#pragma unroll
    for (int kc = 0; kc < 2; ++kc) {
      const bf16_t* pp = &ldsP[wave][lanelo * 64 + kc * 32 + hi * 8];
      v16bf ap = bf_combine(*(const v8bf*)(pp), *(const v8bf*)(pp + 16));
#pragma unroll
      for (int nt = 0; nt < 4; ++nt) {
        const bf16_t* vp = &ldsVt[(nt * 16 + lanelo) * 80 + kc * 32 + hi * 8];
        v16bf bv = bf_combine(*(const v8bf*)(vp), *(const v8bf*)(vp + 16));
        o[nt] = __builtin_amdgcn_wmma_f32_16x16x32_bf16(
            false, ap, false, bv, (short)0, o[nt], false, false);
      }
    }
    __syncthreads();  // protect LDS V tile (and retired K buffer) for next iter
  }

  // ---- finalize: O /= L, store bf16 [B,T,NHD] ----
  float rinv[8];
#pragma unroll
  for (int r = 0; r < 8; ++r) rinv[r] = 1.0f / lrow[r];
#pragma unroll
  for (int nt = 0; nt < 4; ++nt) {
    const int d = nt * 16 + lanelo;
#pragma unroll
    for (int r = 0; r < 8; ++r) {
      const int m = m0 + hi * 8 + r;
      Ao[base + (size_t)m * NHD + d] = (bf16_t)(o[nt][r] * rinv[r]);
    }
  }
}

// ---------------------------------------------------------------------------
// Host-side orchestration
// ---------------------------------------------------------------------------
extern "C" void kernel_launch(void* const* d_in, const int* in_sizes, int n_in,
                              void* d_out, int out_size, void* d_ws, size_t ws_size,
                              hipStream_t stream) {
  (void)in_sizes; (void)n_in; (void)out_size; (void)ws_size;

  const float* inputs_q  = (const float*)d_in[0];
  const float* inputs_kv = (const float*)d_in[1];
  // d_in[2] = boolean causal mask: applied analytically in flash_attn_wmma
  const float* wq     = (const float*)d_in[3];
  const float* wk     = (const float*)d_in[4];
  const float* wv     = (const float*)d_in[5];
  const float* wo     = (const float*)d_in[6];
  const float* q_ln_s = (const float*)d_in[7];
  const float* q_ln_b = (const float*)d_in[8];
  const float* k_ln_s = (const float*)d_in[9];
  const float* k_ln_b = (const float*)d_in[10];
  float* out = (float*)d_out;

  const int M = BB * TT;  // 4096 rows

  char* ws = (char*)d_ws;
  size_t off = 0;
  auto take = [&](size_t bytes) -> char* {
    char* p = ws + off;
    off += (bytes + 255) & ~(size_t)255;
    return p;
  };
  bf16_t* xq_bf  = (bf16_t*)take((size_t)M * EE * 2);
  bf16_t* xkv_bf = (bf16_t*)take((size_t)M * EE * 2);
  bf16_t* wq_bf  = (bf16_t*)take((size_t)NHD * EE * 2);
  bf16_t* wk_bf  = (bf16_t*)take((size_t)NHD * EE * 2);
  bf16_t* wv_bf  = (bf16_t*)take((size_t)NHD * EE * 2);
  bf16_t* wo_bf  = (bf16_t*)take((size_t)EE * NHD * 2);
  float*  q_f32  = (float*) take((size_t)M * NHD * 4);
  float*  k_f32  = (float*) take((size_t)M * NHD * 4);
  bf16_t* qn_bf  = (bf16_t*)take((size_t)M * NHD * 2);
  bf16_t* kn_bf  = (bf16_t*)take((size_t)M * NHD * 2);
  bf16_t* v_bf   = (bf16_t*)take((size_t)M * NHD * 2);
  bf16_t* ao_bf  = (bf16_t*)take((size_t)M * NHD * 2);

  // 1) down-convert activations + weights to bf16
  int n4 = M * EE / 4;
  cvt_f32_to_bf16<<<(n4 + 255) / 256, 256, 0, stream>>>(inputs_q,  xq_bf,  n4);
  cvt_f32_to_bf16<<<(n4 + 255) / 256, 256, 0, stream>>>(inputs_kv, xkv_bf, n4);
  n4 = NHD * EE / 4;
  cvt_f32_to_bf16<<<(n4 + 255) / 256, 256, 0, stream>>>(wq, wq_bf, n4);
  cvt_f32_to_bf16<<<(n4 + 255) / 256, 256, 0, stream>>>(wk, wk_bf, n4);
  cvt_f32_to_bf16<<<(n4 + 255) / 256, 256, 0, stream>>>(wv, wv_bf, n4);
  cvt_f32_to_bf16<<<(n4 + 255) / 256, 256, 0, stream>>>(wo, wo_bf, n4);

  // 2) QKV projections (WMMA); V straight to bf16, Q/K to f32 for LN
  dim3 gp(M / 64, NHD / 64);
  gemm_bf16_wmma<false><<<gp, 128, 0, stream>>>(xq_bf,  wq_bf, q_f32, M, NHD, EE);
  gemm_bf16_wmma<false><<<gp, 128, 0, stream>>>(xkv_bf, wk_bf, k_f32, M, NHD, EE);
  gemm_bf16_wmma<true ><<<gp, 128, 0, stream>>>(xkv_bf, wv_bf, v_bf,  M, NHD, EE);

  // 3) QK LayerNorm -> bf16
  layernorm_to_bf16<<<M, 256, 0, stream>>>(q_f32, q_ln_s, q_ln_b, qn_bf);
  layernorm_to_bf16<<<M, 256, 0, stream>>>(k_f32, k_ln_s, k_ln_b, kn_bf);

  // 4) causal flash attention (WMMA + TDM double-buffered K tiles)
  flash_attn_wmma<<<dim3(BB * HH, TT / 64), 128, 0, stream>>>(qn_bf, kn_bf, v_bf, ao_bf);

  // 5) output projection (WMMA) -> f32 d_out
  dim3 go(M / 64, EE / 64);
  gemm_bf16_wmma<false><<<go, 128, 0, stream>>>(ao_bf, wo_bf, out, M, EE, NHD);
}
